// Model_76862734729788
// MI455X (gfx1250) — matrix operand order repaired and checked by reading, not compiled, not run on hardware
//
#include <hip/hip_runtime.h>
#include <hip/hip_bf16.h>

// ---------------------------------------------------------------------------
// Types for CDNA5 WMMA (wave32): 16x16x32 f16 -> f32 accumulate
// ---------------------------------------------------------------------------
typedef __attribute__((ext_vector_type(16))) _Float16 v16h;
typedef __attribute__((ext_vector_type(8)))  float    v8f;

static __device__ inline _Float16 f2h(float f) { return (_Float16)f; }

// ---------------------------------------------------------------------------
// Generic single-wave WMMA GEMM tile kernel, 16(M) x 32(N) tile per wave.
//   C[m,n] = scale * sum_k A(m,k) * W(n,k) [+ bias[n]] [+ resid[m,n]] [ReLU]
// A(m,k) at A[m*saR + k*saC], W(n,k) at W[n*swN + k*swK].
// blockIdx.z decomposed as (zb = z/zh, zs = z%zh) with separate strides so a
// single launch covers all (batch, head) sub-GEMMs of attention.
// One wave (32 threads) per tile; K must be a multiple of 32, N of 32.
// The A fragment is reused by two WMMAs (two B fragments / accumulators).
// ---------------------------------------------------------------------------
template<int HB, int HR, int RELU>
__global__ __launch_bounds__(32)
void gemm_wmma_k(const float* __restrict__ A, long long aB1, long long aB2, int saR, int saC,
                 const float* __restrict__ W, long long wB1, long long wB2, int swN, int swK,
                 const float* __restrict__ bias,
                 const float* __restrict__ resid,
                 float* __restrict__ C, long long cB1, long long cB2, int ldc,
                 int K, int zh, float scale)
{
    int lane = threadIdx.x;
    int n0 = blockIdx.x << 5;           // 32 output columns per wave
    int m0 = blockIdx.y << 4;
    int z  = blockIdx.z;
    int zb = z / zh;
    int zs = z - zb * zh;
    A += zb * aB1 + zs * aB2;
    W += zb * wB1 + zs * wB2;
    C += zb * cB1 + zs * cB2;

    int r   = m0 + (lane & 15);         // A row for this lane
    int c0  = n0 + (lane & 15);         // first W row (= output column)
    int c1  = c0 + 16;                  // second W row
    int akb = (lane < 16) ? 0 : 16;     // A-fragment K sub-offset (contiguous k = 16h + s)
    int bkb = (lane < 16) ? 0 : 16;     // B-fragment K sub-offset (contiguous k = 16h + s)

    const float* Ar = A + (long long)r  * saR;
    const float* W0 = W + (long long)c0 * swN;
    const float* W1 = W + (long long)c1 * swN;

    v8f acc0 = {0.f, 0.f, 0.f, 0.f, 0.f, 0.f, 0.f, 0.f};
    v8f acc1 = {0.f, 0.f, 0.f, 0.f, 0.f, 0.f, 0.f, 0.f};

    for (int k0 = 0; k0 < K; k0 += 32) {
        // speculative prefetch of next K-tile (global_prefetch_b8; dropped if OOB)
        __builtin_prefetch(Ar + (long long)(k0 + 32 + akb) * saC, 0, 1);
        __builtin_prefetch(W0 + (long long)(k0 + 32 + bkb) * swK, 0, 1);
        __builtin_prefetch(W1 + (long long)(k0 + 32 + bkb) * swK, 0, 1);

        v16h af, bf0, bf1;
#pragma unroll
        for (int j = 0; j < 8; ++j) {
            af[j]     = f2h(Ar[(long long)(k0 + akb + j) * saC]);
            af[j + 8] = f2h(Ar[(long long)(k0 + akb + 8 + j) * saC]);
        }
#pragma unroll
        for (int j = 0; j < 16; ++j) {
            bf0[j] = f2h(W0[(long long)(k0 + bkb + j) * swK]);
            bf1[j] = f2h(W1[(long long)(k0 + bkb + j) * swK]);
        }
        acc0 = __builtin_amdgcn_wmma_f32_16x16x32_f16(
                   false, af, false, bf0, (short)0, acc0, false, false);
        acc1 = __builtin_amdgcn_wmma_f32_16x16x32_f16(
                   false, af, false, bf1, (short)0, acc1, false, false);
    }

    float bv0 = HB ? bias[c0] : 0.f;
    float bv1 = HB ? bias[c1] : 0.f;
    int mb = m0 + ((lane >> 4) << 3);   // D layout: VGPR v holds row m = mb + v
#pragma unroll
    for (int v = 0; v < 8; ++v) {
        int m = mb + v;
        float o0 = acc0[v] * scale + bv0;
        float o1 = acc1[v] * scale + bv1;
        if (HR) {
            o0 += resid[(long long)m * ldc + c0];
            o1 += resid[(long long)m * ldc + c1];
        }
        if (RELU) { o0 = fmaxf(o0, 0.f); o1 = fmaxf(o1, 0.f); }
        C[(long long)m * ldc + c0] = o0;
        C[(long long)m * ldc + c1] = o1;
    }
}

// ---------------------------------------------------------------------------
// Stage 1: periodicity embedding.  x:[B,128,128,257] -> h1:[B,T,1024] where
// channel = ped*128 + band.  One wave per (b,band,t) row; ep_w cached in LDS.
// ---------------------------------------------------------------------------
__global__ __launch_bounds__(256)
void embed_k(const float* __restrict__ x, const float* __restrict__ ep_w,
             const float* __restrict__ ep_b, float* __restrict__ h1)
{
    __shared__ float sw[8 * 257];
    for (int i = threadIdx.x; i < 8 * 257; i += 256) sw[i] = ep_w[i];
    __syncthreads();

    int wid  = threadIdx.x >> 5;
    int lane = threadIdx.x & 31;
    int row  = blockIdx.x * 8 + wid;          // 0 .. 65535
    const float* xr = x + (long long)row * 257;

    float acc[8] = {0.f, 0.f, 0.f, 0.f, 0.f, 0.f, 0.f, 0.f};
#pragma unroll
    for (int i = 0; i < 8; ++i) {
        float xv = xr[i * 32 + lane];
#pragma unroll
        for (int p = 0; p < 8; ++p) acc[p] += xv * sw[p * 257 + i * 32 + lane];
    }
    if (lane == 0) {
        float xv = xr[256];
#pragma unroll
        for (int p = 0; p < 8; ++p) acc[p] += xv * sw[p * 257 + 256];
    }
#pragma unroll
    for (int off = 16; off >= 1; off >>= 1) {
#pragma unroll
        for (int p = 0; p < 8; ++p) acc[p] += __shfl_xor(acc[p], off, 32);
    }
    if (lane < 8) {
        int b = row >> 14, band = (row >> 7) & 127, t = row & 127;
        h1[((long long)(b * 128 + t)) * 1024 + lane * 128 + band] = acc[lane] + ep_b[lane];
    }
}

// ---------------------------------------------------------------------------
// Positional-feature projection folded into a per-(b,t,o) additive term:
//   pp[(b*128+t)*128 + o] = sum_j down_w[o,128+j]*posfeat[j][t] + down_b[o]
// ---------------------------------------------------------------------------
__global__ __launch_bounds__(256)
void posproj_k(const float* __restrict__ down_w, const float* __restrict__ down_b,
               float* __restrict__ pp)
{
    int id = blockIdx.x * 256 + threadIdx.x;   // 16384 = 128*128
    int o = id >> 7, t = id & 127;
    float pos = -1.f + 2.f * (float)t / 127.f;
    const float* w = down_w + o * 161 + 128;
    float s = w[0] * pos + down_b[o];
    float sc = 3.14159265358979323846f;
#pragma unroll
    for (int k = 0; k < 16; ++k) {
        float a = pos * sc;
        s += w[1 + k] * sinf(a) + w[17 + k] * cosf(a);
        sc *= 2.f;
    }
#pragma unroll
    for (int b = 0; b < 4; ++b) pp[((b * 128 + t) * 128) + o] = s;
}

// ---------------------------------------------------------------------------
// Row LayerNorm over last dim (128).  One block (128 threads) per row.
// ---------------------------------------------------------------------------
__global__ __launch_bounds__(128)
void ln_rows_k(const float* __restrict__ X, const float* __restrict__ g,
               const float* __restrict__ b, float* __restrict__ Y)
{
    __shared__ float sm[128];
    int row = blockIdx.x;
    float v = X[row * 128 + threadIdx.x];
    sm[threadIdx.x] = v; __syncthreads();
    for (int s = 64; s > 0; s >>= 1) {
        if (threadIdx.x < s) sm[threadIdx.x] += sm[threadIdx.x + s];
        __syncthreads();
    }
    float mean = sm[0] * (1.f / 128.f); __syncthreads();
    float d = v - mean;
    sm[threadIdx.x] = d * d; __syncthreads();
    for (int s = 64; s > 0; s >>= 1) {
        if (threadIdx.x < s) sm[threadIdx.x] += sm[threadIdx.x + s];
        __syncthreads();
    }
    float var = sm[0] * (1.f / 128.f);
    Y[row * 128 + threadIdx.x] = d * rsqrtf(var + 1e-5f) * g[threadIdx.x] + b[threadIdx.x];
}

// ---------------------------------------------------------------------------
// Joint LayerNorm over (T, C) = 16384 elements per batch.
// ---------------------------------------------------------------------------
__global__ __launch_bounds__(256)
void final_ln_k(const float* __restrict__ X, const float* __restrict__ g,
                const float* __restrict__ b, float* __restrict__ Y)
{
    __shared__ float sm[256];
    int bb = blockIdx.x;
    const float* x = X + bb * 16384;
    float s = 0.f, s2 = 0.f;
    for (int i = threadIdx.x; i < 16384; i += 256) { float v = x[i]; s += v; s2 += v * v; }
    sm[threadIdx.x] = s; __syncthreads();
    for (int st = 128; st > 0; st >>= 1) {
        if (threadIdx.x < st) sm[threadIdx.x] += sm[threadIdx.x + st];
        __syncthreads();
    }
    float mean = sm[0] * (1.f / 16384.f); __syncthreads();
    sm[threadIdx.x] = s2; __syncthreads();
    for (int st = 128; st > 0; st >>= 1) {
        if (threadIdx.x < st) sm[threadIdx.x] += sm[threadIdx.x + st];
        __syncthreads();
    }
    float var = sm[0] * (1.f / 16384.f) - mean * mean;
    float inv = rsqrtf(var + 1e-5f);
    for (int i = threadIdx.x; i < 16384; i += 256)
        Y[bb * 16384 + i] = (x[i] - mean) * inv * g[i] + b[i];
}

// ---------------------------------------------------------------------------
// Softmax over last dim (128); one block per attention row.
// ---------------------------------------------------------------------------
__global__ __launch_bounds__(128)
void softmax_k(float* __restrict__ S)
{
    __shared__ float sm[128];
    int row = blockIdx.x;
    float v = S[row * 128 + threadIdx.x];
    sm[threadIdx.x] = v; __syncthreads();
    for (int s = 64; s > 0; s >>= 1) {
        if (threadIdx.x < s) sm[threadIdx.x] = fmaxf(sm[threadIdx.x], sm[threadIdx.x + s]);
        __syncthreads();
    }
    float mx = sm[0]; __syncthreads();
    float e = __expf(v - mx);
    sm[threadIdx.x] = e; __syncthreads();
    for (int s = 64; s > 0; s >>= 1) {
        if (threadIdx.x < s) sm[threadIdx.x] += sm[threadIdx.x + s];
        __syncthreads();
    }
    S[row * 128 + threadIdx.x] = e / sm[0];
}

// ---------------------------------------------------------------------------
// Top-128 selection per batch via 32-step binary search on sortable key.
// up layout: [b][t][512]; encoding flat index = o*128 + t.
// ---------------------------------------------------------------------------
static __device__ inline unsigned fkey(float f) {
    unsigned u = __builtin_bit_cast(unsigned, f);
    return (u & 0x80000000u) ? ~u : (u | 0x80000000u);
}

__global__ __launch_bounds__(256)
void topk_k(const float* __restrict__ up, float* __restrict__ enc,
            int* __restrict__ kidx, float* __restrict__ kval)
{
    __shared__ unsigned sred[256];
    __shared__ int s_gt, s_eq;
    int b = blockIdx.x;
    const float* U = up + (long long)b * 65536;

    unsigned lo = 0u, hi = 0xFFFFFFFFu;
    for (int it = 0; it < 32 && lo < hi; ++it) {
        unsigned mid = lo + (unsigned)((((unsigned long long)(hi - lo)) + 1ull) >> 1);
        unsigned cth = 0;
        for (int i = threadIdx.x; i < 65536; i += 256)
            cth += (fkey(U[i]) >= mid) ? 1u : 0u;
        sred[threadIdx.x] = cth; __syncthreads();
        for (int s = 128; s > 0; s >>= 1) {
            if (threadIdx.x < s) sred[threadIdx.x] += sred[threadIdx.x + s];
            __syncthreads();
        }
        unsigned cnt = sred[0]; __syncthreads();
        if (cnt >= 128u) lo = mid; else hi = mid - 1u;
    }
    unsigned tau = lo;

    unsigned cth = 0;
    for (int i = threadIdx.x; i < 65536; i += 256)
        cth += (fkey(U[i]) > tau) ? 1u : 0u;
    sred[threadIdx.x] = cth; __syncthreads();
    for (int s = 128; s > 0; s >>= 1) {
        if (threadIdx.x < s) sred[threadIdx.x] += sred[threadIdx.x + s];
        __syncthreads();
    }
    int cgt = (int)sred[0];
    if (threadIdx.x == 0) { s_gt = 0; s_eq = 0; }
    __syncthreads();

    for (int i = threadIdx.x; i < 65536; i += 256) {
        float v = U[i];
        unsigned k = fkey(v);
        int slot = -1;
        if (k > tau) {
            slot = atomicAdd(&s_gt, 1);
        } else if (k == tau) {
            int e = atomicAdd(&s_eq, 1);
            if (cgt + e < 128) slot = cgt + e;
        }
        if (slot >= 0) {
            int t = i >> 9, o = i & 511;
            int fl = o * 128 + t;
            kidx[b * 128 + slot] = fl;
            kval[b * 128 + slot] = v;
            enc[(long long)b * 65536 + fl] = v;
        }
    }
}

// ---------------------------------------------------------------------------
// Sparse circular convolution: each kept (c,t,v) adds v*atoms[c,0..511] at
// signal positions (t*256 + j) mod 32768. Replaces the dense FFT convolve.
// ---------------------------------------------------------------------------
__global__ __launch_bounds__(128)
void sconv_k(const int* __restrict__ kidx, const float* __restrict__ kval,
             const float* __restrict__ atoms, float* __restrict__ sig)
{
    int s = blockIdx.x;          // 0 .. 4*128-1
    int b = s >> 7;
    int fl = kidx[s];
    float v = kval[s];
    int o = fl >> 7, t = fl & 127;
    int p = t << 8;              // ratio = 32768/128 = 256
    const float* at = atoms + o * 512;
    float* sg = sig + (long long)b * 32768;
    for (int j = threadIdx.x; j < 512; j += 128)
        atomicAdd(sg + ((p + j) & 32767), v * at[j]);
}

__global__ __launch_bounds__(256)
void zero_k(float* __restrict__ p, int n)
{
    int i = blockIdx.x * 256 + threadIdx.x;
    if (i < n) p[i] = 0.f;
}

// ---------------------------------------------------------------------------
// Host-side launch helper: dispatch to the right template instantiation so
// the device epilogue has no runtime branches.
// ---------------------------------------------------------------------------
static inline void gemm(hipStream_t st,
                        const float* A, long long aB1, long long aB2, int saR, int saC,
                        const float* W, long long wB1, long long wB2, int swN, int swK,
                        const float* bias, const float* resid,
                        float* C, long long cB1, long long cB2, int ldc,
                        int M, int N, int K, int Z, int zh, float scale,
                        int hb, int hr, int rl)
{
    dim3 g(N / 32, M / 16, Z);
#define GEMM_LAUNCH(HB, HR, RL)                                                    \
    gemm_wmma_k<HB, HR, RL><<<g, 32, 0, st>>>(A, aB1, aB2, saR, saC,               \
                                              W, wB1, wB2, swN, swK, bias, resid,  \
                                              C, cB1, cB2, ldc, K, zh, scale)
    int sel = (hb ? 4 : 0) | (hr ? 2 : 0) | (rl ? 1 : 0);
    switch (sel) {
        case 0: GEMM_LAUNCH(0, 0, 0); break;
        case 1: GEMM_LAUNCH(0, 0, 1); break;
        case 2: GEMM_LAUNCH(0, 1, 0); break;
        case 3: GEMM_LAUNCH(0, 1, 1); break;
        case 4: GEMM_LAUNCH(1, 0, 0); break;
        case 5: GEMM_LAUNCH(1, 0, 1); break;
        case 6: GEMM_LAUNCH(1, 1, 0); break;
        default: GEMM_LAUNCH(1, 1, 1); break;
    }
#undef GEMM_LAUNCH
}

extern "C" void kernel_launch(void* const* d_in, const int* in_sizes, int n_in,
                              void* d_out, int out_size, void* d_ws, size_t ws_size,
                              hipStream_t stream)
{
    (void)in_sizes; (void)n_in; (void)ws_size; (void)out_size;

    const float* x        = (const float*)d_in[0];
    const float* ep_w     = (const float*)d_in[1];
    const float* ep_b     = (const float*)d_in[2];
    const float* reduce_w = (const float*)d_in[3];
    const float* reduce_b = (const float*)d_in[4];
    const float* down_w   = (const float*)d_in[5];
    const float* down_b   = (const float*)d_in[6];
    const float* fln_g    = (const float*)d_in[79];
    const float* fln_b    = (const float*)d_in[80];
    const float* up_w     = (const float*)d_in[81];
    const float* up_b     = (const float*)d_in[82];
    const float* atoms    = (const float*)d_in[83];

    float* ws  = (float*)d_ws;
    float* h1  = ws;                 // [B,T,1024]       524288
    float* h2  = h1  + 524288;       // [B*T,128]         65536
    float* pp  = h2  + 65536;        // [B*T,128]         65536
    float* xb  = pp  + 65536;        // [B*T,128]         65536
    float* qkv = xb  + 65536;        // [B*T,384]        196608
    float* sc  = qkv + 196608;       // [B,H,T,T]        262144
    float* ao  = sc  + 262144;       // [B*T,128]         65536
    float* t1  = ao  + 65536;        // [B*T,128]         65536
    float* t2  = t1  + 65536;        // [B*T,128]         65536
    float* xb2 = t2  + 65536;        // [B*T,128]         65536
    float* upb = xb2 + 65536;        // [B*T,512]        262144
    float* kval = upb + 262144;      // [B,128]             512
    int*   kidx = (int*)(kval + 512);// [B,128]             512

    float* out = (float*)d_out;
    float* sig = out;                // [B,1,32768] = 131072
    float* enc = out + 4 * 32768;    // [B,512,128] = 262144

    // zero outputs (poisoned by harness; atomics/scatter need clean base)
    zero_k<<<(393216 + 255) / 256, 256, 0, stream>>>(out, 393216);

    // 1) periodicity embedding (memory-bound: streams the 67MB input once)
    embed_k<<<8192, 256, 0, stream>>>(x, ep_w, ep_b, h1);

    // 2) reduce 1x1 conv: [512,1024] x [128,1024]^T
    gemm(stream, h1, 0, 0, 1024, 1, reduce_w, 0, 0, 1024, 1,
         reduce_b, nullptr, h2, 0, 0, 128, 512, 128, 1024, 1, 1, 1.f, 1, 0, 0);

    // 3) positional features folded into additive term, then down conv
    posproj_k<<<64, 256, 0, stream>>>(down_w, down_b, pp);
    gemm(stream, h2, 0, 0, 128, 1, down_w, 0, 0, 161, 1,
         nullptr, pp, xb, 0, 0, 128, 512, 128, 128, 1, 1, 1.f, 0, 1, 0);

    // 4) six transformer encoder layers (post-norm, ffn=relu, 4 heads of 32)
    const float inv_sqrt_hd = 0.17677669529663687f; // 1/sqrt(32)
    for (int l = 0; l < 6; ++l) {
        const float* in_w  = (const float*)d_in[7 + l * 12 + 0];
        const float* in_b  = (const float*)d_in[7 + l * 12 + 1];
        const float* out_w = (const float*)d_in[7 + l * 12 + 2];
        const float* out_b = (const float*)d_in[7 + l * 12 + 3];
        const float* ff1_w = (const float*)d_in[7 + l * 12 + 4];
        const float* ff1_b = (const float*)d_in[7 + l * 12 + 5];
        const float* ff2_w = (const float*)d_in[7 + l * 12 + 6];
        const float* ff2_b = (const float*)d_in[7 + l * 12 + 7];
        const float* ln1_g = (const float*)d_in[7 + l * 12 + 8];
        const float* ln1_b = (const float*)d_in[7 + l * 12 + 9];
        const float* ln2_g = (const float*)d_in[7 + l * 12 + 10];
        const float* ln2_b = (const float*)d_in[7 + l * 12 + 11];

        // QKV projection: [512,128] x [384,128]^T
        gemm(stream, xb, 0, 0, 128, 1, in_w, 0, 0, 128, 1,
             in_b, nullptr, qkv, 0, 0, 384, 512, 384, 128, 1, 1, 1.f, 1, 0, 0);

        // scores[b,h] = Q K^T / sqrt(hd) : z over (b,h), strides decomposed
        gemm(stream, qkv, 49152, 32, 384, 1, qkv + 128, 49152, 32, 384, 1,
             nullptr, nullptr, sc, 65536, 16384, 128,
             128, 128, 32, 16, 4, inv_sqrt_hd, 0, 0, 0);

        softmax_k<<<2048, 128, 0, stream>>>(sc);

        // att x V: W(n,k) = V[k,n] via swN=1, swK=384 (no transpose pass)
        gemm(stream, sc, 65536, 16384, 128, 1, qkv + 256, 49152, 32, 1, 384,
             nullptr, nullptr, ao, 16384, 32, 128,
             128, 32, 128, 16, 4, 1.f, 0, 0, 0);

        // out proj + residual, then LN1
        gemm(stream, ao, 0, 0, 128, 1, out_w, 0, 0, 128, 1,
             out_b, xb, t1, 0, 0, 128, 512, 128, 128, 1, 1, 1.f, 1, 1, 0);
        ln_rows_k<<<512, 128, 0, stream>>>(t1, ln1_g, ln1_b, xb2);

        // FFN: relu(x W1^T + b1) W2^T + b2 + residual, then LN2
        gemm(stream, xb2, 0, 0, 128, 1, ff1_w, 0, 0, 128, 1,
             ff1_b, nullptr, t1, 0, 0, 128, 512, 128, 128, 1, 1, 1.f, 1, 0, 1);
        gemm(stream, t1, 0, 0, 128, 1, ff2_w, 0, 0, 128, 1,
             ff2_b, xb2, t2, 0, 0, 128, 512, 128, 128, 1, 1, 1.f, 1, 1, 0);
        ln_rows_k<<<512, 128, 0, stream>>>(t2, ln2_g, ln2_b, xb);
    }

    // 5) joint LayerNorm over (T,C) per batch
    final_ln_k<<<4, 256, 0, stream>>>(xb, fln_g, fln_b, xb2);

    // 6) up 1x1 conv: [512,128] x [512,128]^T -> [B*T,512]
    gemm(stream, xb2, 0, 0, 128, 1, up_w, 0, 0, 128, 1,
         up_b, nullptr, upb, 0, 0, 512, 512, 512, 128, 1, 1, 1.f, 1, 0, 0);

    // 7) top-128 per batch -> encoding output + kept list
    topk_k<<<4, 256, 0, stream>>>(upb, enc, kidx, kval);

    // 8) sparse circular convolution with atoms, summed over channels
    sconv_k<<<512, 128, 0, stream>>>(kidx, kval, atoms, sig);
}
